// HRViTBlock_31190052504217
// MI455X (gfx1250) — compile-verified, matrix operand order checked by reading, not run on hardware
//
#include <hip/hip_runtime.h>
#include <math.h>
#include <stdint.h>

// ---------------------------------------------------------------------------
// HRViT block for MI455X (gfx1250): bf16 WMMA GEMMs (transposed-B operands),
// LDS-resident window attention with async global->LDS V staging.
// ---------------------------------------------------------------------------

typedef __attribute__((ext_vector_type(16))) __bf16 v16bf;
typedef __attribute__((ext_vector_type(8)))  __bf16 v8bf;
typedef __attribute__((ext_vector_type(8)))  float  v8f;

#define BATCH 4
#define NTOK  4096            // 64*64 tokens per image
#define ROWS  (BATCH * NTOK)  // 16384
#define CDIM  256
#define HIDD  1024

__device__ __forceinline__ float hswish(float x) {
    float t = x + 3.0f;
    t = fminf(fmaxf(t, 0.0f), 6.0f);
    return x * t * (1.0f / 6.0f);
}
__device__ __forceinline__ float gelu_exact(float x) {
    return 0.5f * x * (1.0f + erff(x * 0.70710678118654752f));
}
__device__ __forceinline__ __bf16 f2bf(float x) { return (__bf16)x; }

// K index packing for 16-bit WMMA A/B operands (16x32 / 32x16), wave32.
__device__ __forceinline__ int wmma_k(int j, int half) {
    return (j < 4) ? (half * 8 + 2 * j) : (16 + half * 8 + 2 * (j - 4));
}

// Assemble one lane's 16xbf16 WMMA operand from the two contiguous 8-element
// runs the layout calls for: [base, base+8) and [base+16, base+24).
// p must already point at (row, k0 + half*8).
__device__ __forceinline__ v16bf load_op32(const __bf16* p) {
    v8bf lo = *(const v8bf*)p;
    v8bf hi = *(const v8bf*)(p + 16);
    return __builtin_shufflevector(lo, hi, 0, 1, 2, 3, 4, 5, 6, 7,
                                   8, 9, 10, 11, 12, 13, 14, 15);
}

// ---------------------------------------------------------------------------
// bf16 GEMM with pre-transposed B:  C(MxN,f32) = A(MxK) * Bt(NxK)^T + bias
// One 16x64 tile per wave (4 WMMAs / k-step, A reused), 8 waves / block.
// ---------------------------------------------------------------------------
template <bool RESID>
__global__ __launch_bounds__(256)
void gemm_bf16_kernel(const __bf16* __restrict__ A, int lda,
                      const __bf16* __restrict__ Bt,   // N x K row-major
                      const float* __restrict__ bias,
                      const float* __restrict__ resid,
                      float* __restrict__ C, int ldc,
                      int M, int N, int K) {
    int wave = threadIdx.x >> 5;
    int lane = threadIdx.x & 31;
    int m = lane & 15, half = lane >> 4;
    int tiles_n = N >> 6;
    int tile = blockIdx.x * 8 + wave;
    if (tile >= (M >> 4) * tiles_n) return;          // wave-uniform exit
    int tm = tile / tiles_n, tn = tile % tiles_n;

    const __bf16* Arow = A + (size_t)(tm * 16 + m) * lda + half * 8;
    const __bf16* Brow = Bt + (size_t)(tn * 64 + m) * K + half * 8;
    v8f acc0 = {}, acc1 = {}, acc2 = {}, acc3 = {};
    for (int k0 = 0; k0 < K; k0 += 32) {
        v16bf a  = load_op32(Arow + k0);
        v16bf b0 = load_op32(Brow + k0);
        v16bf b1 = load_op32(Brow + (size_t)16 * K + k0);
        v16bf b2 = load_op32(Brow + (size_t)32 * K + k0);
        v16bf b3 = load_op32(Brow + (size_t)48 * K + k0);
        acc0 = __builtin_amdgcn_wmma_f32_16x16x32_bf16(false, a, false, b0, (short)0, acc0, false, false);
        acc1 = __builtin_amdgcn_wmma_f32_16x16x32_bf16(false, a, false, b1, (short)0, acc1, false, false);
        acc2 = __builtin_amdgcn_wmma_f32_16x16x32_bf16(false, a, false, b2, (short)0, acc2, false, false);
        acc3 = __builtin_amdgcn_wmma_f32_16x16x32_bf16(false, a, false, b3, (short)0, acc3, false, false);
    }
    int coln = tn * 64 + m;
    float bi0 = bias[coln], bi1 = bias[coln + 16], bi2 = bias[coln + 32], bi3 = bias[coln + 48];
#pragma unroll
    for (int r = 0; r < 8; ++r) {
        int row = tm * 16 + r + 8 * half;
        size_t o = (size_t)row * ldc + coln;
        float v0 = acc0[r] + bi0, v1 = acc1[r] + bi1;
        float v2 = acc2[r] + bi2, v3 = acc3[r] + bi3;
        if (RESID) {
            v0 += resid[o]; v1 += resid[o + 16];
            v2 += resid[o + 32]; v3 += resid[o + 48];
        }
        C[o] = v0; C[o + 16] = v1; C[o + 32] = v2; C[o + 48] = v3;
    }
}

// ---------------------------------------------------------------------------
// LayerNorm over last dim (256) -> bf16. One block (256 thr) per row.
// ---------------------------------------------------------------------------
__global__ __launch_bounds__(256)
void layernorm_bf16_kernel(const float* __restrict__ x,
                           const float* __restrict__ w,
                           const float* __restrict__ b,
                           __bf16* __restrict__ out) {
    __shared__ float sbuf[256];
    int row = blockIdx.x, tid = threadIdx.x;
    float v = x[(size_t)row * CDIM + tid];
    sbuf[tid] = v; __syncthreads();
    for (int s = 128; s > 0; s >>= 1) { if (tid < s) sbuf[tid] += sbuf[tid + s]; __syncthreads(); }
    float mu = sbuf[0] * (1.0f / CDIM);
    __syncthreads();
    float d = v - mu;
    sbuf[tid] = d * d; __syncthreads();
    for (int s = 128; s > 0; s >>= 1) { if (tid < s) sbuf[tid] += sbuf[tid + s]; __syncthreads(); }
    float var = sbuf[0] * (1.0f / CDIM);
    out[(size_t)row * CDIM + tid] = f2bf(d * rsqrtf(var + 1e-5f) * w[tid] + b[tid]);
}

// Split qv; emit q,v as bf16 and hardswish(v) image (B,C,H,W).
__global__ __launch_bounds__(256)
void post_qkv_kernel(const float* __restrict__ qv,
                     __bf16* __restrict__ qb, __bf16* __restrict__ vb,
                     float* __restrict__ hsv) {
    int idx = blockIdx.x * 256 + threadIdx.x;  // row*256 + c
    int row = idx >> 8, c = idx & 255;
    float q = qv[(size_t)row * 512 + c];
    float v = qv[(size_t)row * 512 + 256 + c];
    qb[idx] = f2bf(q);
    vb[idx] = f2bf(v);
    int b = row >> 12, pos = row & 4095;
    hsv[(((size_t)(b * 256 + c)) << 12) + pos] = hswish(v);
}

// 3x3 depthwise conv on hsv image -> (row,c) layout f32.
__global__ __launch_bounds__(256)
void dwconv_v_kernel(const float* __restrict__ img, const float* __restrict__ wt,
                     const float* __restrict__ bias, float* __restrict__ outp) {
    int idx = blockIdx.x * 256 + threadIdx.x;  // (b*256+c)*4096 + pos
    int pos = idx & 4095, c = (idx >> 12) & 255, b = idx >> 20;
    int h = pos >> 6, w = pos & 63;
    const float* ip = img + ((size_t)(b * 256 + c) << 12);
    float acc = bias[c];
#pragma unroll
    for (int dy = -1; dy <= 1; ++dy)
#pragma unroll
        for (int dx = -1; dx <= 1; ++dx) {
            int hy = h + dy, wx = w + dx;
            if ((unsigned)hy < 64u && (unsigned)wx < 64u)
                acc += ip[hy * 64 + wx] * wt[c * 9 + (dy + 1) * 3 + (dx + 1)];
        }
    outp[((size_t)(b * 4096 + pos)) * 256 + c] = acc;
}

// ---------------------------------------------------------------------------
// Cross-shaped window attention. Block = 2 waves (64 thr) on a 16-row strip
// of one (batch, stripe, head) window: 512 tokens, d=32.
//   stage : V window (512x32 bf16) -> LDS via async global->LDS b128
//   phase1: S strip (16x512) via WMMA into LDS
//   phase2: row softmax in LDS
//   phase3: O (16x32) via WMMA over K=512 (operands from LDS)
// ---------------------------------------------------------------------------
__global__ __launch_bounds__(64)
void attn_kernel(const __bf16* __restrict__ qb, const __bf16* __restrict__ vb,
                 float* __restrict__ outp, int horizontal) {
    __shared__ float  S[16][512];      // 32 KB scores/probs
    __shared__ __bf16 Vw[512][32];     // 32 KB staged V window
    __shared__ float  red[16][4];

    int tid = threadIdx.x;
    int wave = tid >> 5, lane = tid & 31;
    int m = lane & 15, half = lane >> 4;
    int rb = blockIdx.x;               // row-block 0..31
    int by = blockIdx.y;               // b*32 + g*4 + hh
    int b = by >> 5, rem = by & 31, g = rem >> 2, hh = rem & 3;
    int cbase = (horizontal ? 0 : 128) + hh * 32;

    auto bn_of = [&](int n) -> int {   // local window token -> global row
        int a1 = n >> 6, a0 = n & 63;
        int hpos, wpos;
        if (horizontal) { hpos = g * 8 + a1; wpos = a0; }
        else            { wpos = g * 8 + a1; hpos = a0; }
        return (b << 12) + (hpos << 6) + wpos;
    };

    // ---- stage V window into LDS with async copies (ASYNCcnt path) ----
    for (int n = tid; n < 512; n += 64) {
        const __bf16* src = vb + (size_t)bn_of(n) * 256 + cbase;   // 64 contiguous bytes
        uint32_t dst = (uint32_t)(uintptr_t)&Vw[n][0];
#pragma unroll
        for (int q4 = 0; q4 < 4; ++q4) {
            unsigned long long ga = (unsigned long long)(uintptr_t)(src + q4 * 8);
            uint32_t la = dst + q4 * 16;
            asm volatile("global_load_async_to_lds_b128 %0, %1, off"
                         :: "v"(la), "v"(ga) : "memory");
        }
    }
    asm volatile("s_wait_asynccnt 0x0" ::: "memory");
    __syncthreads();

    int m0 = rb * 16;
    // Q operand (16x32 bf16), fixed for the strip.
    v16bf qa = load_op32(qb + (size_t)bn_of(m0 + m) * 256 + cbase + half * 8);

    const float scale = 0.17677669529663687f;   // 1/sqrt(32)
    for (int nt = wave; nt < 32; nt += 2) {     // 16 col-tiles per wave
        v16bf kbm = load_op32(&Vw[nt * 16 + m][half * 8]);
        v8f acc = {};
        acc = __builtin_amdgcn_wmma_f32_16x16x32_bf16(false, qa, false, kbm,
                                                      (short)0, acc, false, false);
#pragma unroll
        for (int r = 0; r < 8; ++r)
            S[r + 8 * half][nt * 16 + m] = acc[r] * scale;
    }
    __syncthreads();

    // softmax: 4 threads per row
    {
        int row = tid >> 2, sub = tid & 3;
        float mx = -1e30f;
        for (int c = sub; c < 512; c += 4) mx = fmaxf(mx, S[row][c]);
        red[row][sub] = mx; __syncthreads();
        mx = fmaxf(fmaxf(red[row][0], red[row][1]), fmaxf(red[row][2], red[row][3]));
        __syncthreads();
        float sm = 0.0f;
        for (int c = sub; c < 512; c += 4) {
            float e = __expf(S[row][c] - mx);
            S[row][c] = e; sm += e;
        }
        red[row][sub] = sm; __syncthreads();
        sm = red[row][0] + red[row][1] + red[row][2] + red[row][3];
        float inv = 1.0f / sm;
        for (int c = sub; c < 512; c += 4) S[row][c] *= inv;
    }
    __syncthreads();

    // O = P @ V : wave w computes d-cols [w*16, w*16+16), K = 512.
    v8f oacc = {};
    int ntile = wave;
    for (int kt = 0; kt < 16; ++kt) {
        v16bf pa, vbm;
#pragma unroll
        for (int j = 0; j < 8; ++j) {
            int kk = wmma_k(j, half);
            pa[2 * j]      = f2bf(S[m][kt * 32 + kk]);
            pa[2 * j + 1]  = f2bf(S[m][kt * 32 + kk + 1]);
            vbm[2 * j]     = Vw[kt * 32 + kk][ntile * 16 + m];
            vbm[2 * j + 1] = Vw[kt * 32 + kk + 1][ntile * 16 + m];
        }
        oacc = __builtin_amdgcn_wmma_f32_16x16x32_bf16(false, pa, false, vbm,
                                                       (short)0, oacc, false, false);
    }
#pragma unroll
    for (int r = 0; r < 8; ++r) {
        int orow = bn_of(m0 + r + 8 * half);
        outp[(size_t)orow * 256 + cbase + ntile * 16 + m] = oacc[r];
    }
}

// a = hardswish(attn + vconv) -> bf16 for Wout GEMM
__global__ __launch_bounds__(256)
void combine_kernel(const float* __restrict__ attn, const float* __restrict__ vconv,
                    __bf16* __restrict__ ab) {
    int idx = blockIdx.x * 256 + threadIdx.x;
    ab[idx] = f2bf(hswish(attn[idx] + vconv[idx]));
}

// Training-mode BatchNorm batch statistics over 16384 rows per channel.
__global__ __launch_bounds__(256)
void bnstats_kernel(const float* __restrict__ a, float* __restrict__ mu,
                    float* __restrict__ rvar) {
    __shared__ float s1[256], s2[256];
    int c = blockIdx.x, tid = threadIdx.x;
    float sum = 0.0f, sq = 0.0f;
    for (int r = tid; r < ROWS; r += 256) {
        float v = a[(size_t)r * CDIM + c];
        sum += v; sq += v * v;
    }
    s1[tid] = sum; s2[tid] = sq; __syncthreads();
    for (int s = 128; s > 0; s >>= 1) {
        if (tid < s) { s1[tid] += s1[tid + s]; s2[tid] += s2[tid + s]; }
        __syncthreads();
    }
    if (tid == 0) {
        float mm = s1[0] * (1.0f / ROWS);
        mu[c] = mm;
        rvar[c] = rsqrtf(s2[0] * (1.0f / ROWS) - mm * mm + 1e-5f);
    }
}

// BN-apply + DES shortcut (two 16x16 matmuls per row in LDS) + residual.
__global__ __launch_bounds__(256)
void bn_des_res_kernel(const float* __restrict__ xin, const float* __restrict__ aout,
                       const float* __restrict__ mu, const float* __restrict__ rvar,
                       const float* __restrict__ g, const float* __restrict__ bb,
                       const float* __restrict__ wr, const float* __restrict__ br,
                       const float* __restrict__ wl, const float* __restrict__ bl,
                       float* __restrict__ x1) {
    __shared__ float sres[256];
    __shared__ float z[16 * 17];
    int row = blockIdx.x, tid = threadIdx.x;
    float rv = xin[(size_t)row * 256 + tid];
    sres[tid] = rv; __syncthreads();
    int k = tid >> 4, p = tid & 15;           // y1[k][p]
    float y1 = br[p];
#pragma unroll
    for (int pp = 0; pp < 16; ++pp) y1 += sres[k * 16 + pp] * wr[pp * 16 + p];
    z[p * 17 + k] = gelu_exact(y1);           // z[p][k]
    __syncthreads();
    int p2 = tid >> 4, kk = tid & 15;         // y2[p2][kk]
    float y2 = bl[kk];
#pragma unroll
    for (int k2 = 0; k2 < 16; ++k2) y2 += z[p2 * 17 + k2] * wl[k2 * 16 + kk];
    int c = kk * 16 + p2;                     // back to channel index
    float an = (aout[(size_t)row * 256 + c] - mu[c]) * rvar[c] * g[c] + bb[c];
    x1[(size_t)row * 256 + c] = an + y2 + sres[c];
}

// MixCFN depthwise convs (3x3 first 512 ch, 5x5 last 512) + GELU -> bf16.
__global__ __launch_bounds__(256)
void mixconv_kernel(const float* __restrict__ h1,
                    const float* __restrict__ w3, const float* __restrict__ b3,
                    const float* __restrict__ w5, const float* __restrict__ b5,
                    __bf16* __restrict__ hc) {
    int idx = blockIdx.x * 256 + threadIdx.x;  // (b*1024+c)*4096 + pos
    int pos = idx & 4095, c = (idx >> 12) & 1023, b = idx >> 22;
    int h = pos >> 6, w = pos & 63;
    const float* base = h1 + (size_t)b * 4096 * 1024 + c;
    float acc;
    if (c < 512) {
        acc = b3[c];
#pragma unroll
        for (int dy = -1; dy <= 1; ++dy)
#pragma unroll
            for (int dx = -1; dx <= 1; ++dx) {
                int hy = h + dy, wx = w + dx;
                if ((unsigned)hy < 64u && (unsigned)wx < 64u)
                    acc += base[(size_t)(hy * 64 + wx) * 1024] * w3[c * 9 + (dy + 1) * 3 + (dx + 1)];
            }
    } else {
        int c5 = c - 512;
        acc = b5[c5];
#pragma unroll
        for (int dy = -2; dy <= 2; ++dy)
#pragma unroll
            for (int dx = -2; dx <= 2; ++dx) {
                int hy = h + dy, wx = w + dx;
                if ((unsigned)hy < 64u && (unsigned)wx < 64u)
                    acc += base[(size_t)(hy * 64 + wx) * 1024] * w5[c5 * 25 + (dy + 2) * 5 + (dx + 2)];
            }
    }
    hc[(size_t)(b * 4096 + pos) * 1024 + c] = f2bf(gelu_exact(acc));
}

// Transpose-convert a KxN f32 weight into NxK bf16 (WMMA B operand layout).
__global__ __launch_bounds__(256)
void cvt_transpose_bf16_kernel(const float* __restrict__ in, __bf16* __restrict__ out,
                               int K, int N) {
    int idx = blockIdx.x * 256 + threadIdx.x;
    if (idx >= K * N) return;
    int n = idx / K, k = idx - n * K;
    out[idx] = f2bf(in[(size_t)k * N + n]);
}

// ---------------------------------------------------------------------------
extern "C" void kernel_launch(void* const* d_in, const int* in_sizes, int n_in,
                              void* d_out, int out_size, void* d_ws, size_t ws_size,
                              hipStream_t stream) {
    const float* x      = (const float*)d_in[0];
    const float* ln1_w  = (const float*)d_in[3];
    const float* ln1_b  = (const float*)d_in[4];
    const float* Wqkv   = (const float*)d_in[5];
    const float* bqkv   = (const float*)d_in[6];
    const float* convpw = (const float*)d_in[7];
    const float* convpb = (const float*)d_in[8];
    const float* Wout   = (const float*)d_in[9];
    const float* bout   = (const float*)d_in[10];
    const float* bn_g   = (const float*)d_in[11];
    const float* bn_b   = (const float*)d_in[12];
    const float* des_wr = (const float*)d_in[13];
    const float* des_br = (const float*)d_in[14];
    const float* des_wl = (const float*)d_in[15];
    const float* des_bl = (const float*)d_in[16];
    const float* ln2_w  = (const float*)d_in[17];
    const float* ln2_b  = (const float*)d_in[18];
    const float* fc1_w  = (const float*)d_in[19];
    const float* fc1_b  = (const float*)d_in[20];
    const float* mix3_w = (const float*)d_in[21];
    const float* mix3_b = (const float*)d_in[22];
    const float* mix5_w = (const float*)d_in[23];
    const float* mix5_b = (const float*)d_in[24];
    const float* fc2_w  = (const float*)d_in[25];
    const float* fc2_b  = (const float*)d_in[26];
    float* out = (float*)d_out;

    // bump allocator over d_ws
    char* p = (char*)d_ws;
    auto take = [&](size_t bytes) -> char* {
        char* r = p;
        p += (bytes + 255) & ~(size_t)255;
        return r;
    };
    __bf16* wqkv_t  = (__bf16*)take((size_t)512 * 256 * 2);    // N x K
    __bf16* wout_t  = (__bf16*)take((size_t)256 * 256 * 2);
    __bf16* fc1_t   = (__bf16*)take((size_t)1024 * 256 * 2);
    __bf16* fc2_t   = (__bf16*)take((size_t)256 * 1024 * 2);
    __bf16* xn_bf   = (__bf16*)take((size_t)ROWS * CDIM * 2);
    float*  qv      = (float*)take((size_t)ROWS * 512 * 4);
    __bf16* q_bf    = (__bf16*)take((size_t)ROWS * CDIM * 2);
    __bf16* v_bf    = (__bf16*)take((size_t)ROWS * CDIM * 2);
    float*  hsv     = (float*)take((size_t)ROWS * CDIM * 4);
    float*  vconv   = (float*)take((size_t)ROWS * CDIM * 4);
    float*  attn    = (float*)take((size_t)ROWS * CDIM * 4);
    __bf16* a_bf    = (__bf16*)take((size_t)ROWS * CDIM * 2);
    float*  aout    = (float*)take((size_t)ROWS * CDIM * 4);
    float*  mu      = (float*)take(256 * 4);
    float*  rvar    = (float*)take(256 * 4);
    float*  x1      = (float*)take((size_t)ROWS * CDIM * 4);
    __bf16* xn2_bf  = (__bf16*)take((size_t)ROWS * CDIM * 2);
    float*  h1      = (float*)take((size_t)ROWS * HIDD * 4);
    __bf16* hc_bf   = (__bf16*)take((size_t)ROWS * HIDD * 2);

    // 0) weight conversion + transpose to N x K bf16
    cvt_transpose_bf16_kernel<<<512, 256, 0, stream>>>(Wqkv, wqkv_t, 256, 512);
    cvt_transpose_bf16_kernel<<<256, 256, 0, stream>>>(Wout, wout_t, 256, 256);
    cvt_transpose_bf16_kernel<<<1024, 256, 0, stream>>>(fc1_w, fc1_t, 256, 1024);
    cvt_transpose_bf16_kernel<<<1024, 256, 0, stream>>>(fc2_w, fc2_t, 1024, 256);

    // 1) LN1
    layernorm_bf16_kernel<<<ROWS, 256, 0, stream>>>(x, ln1_w, ln1_b, xn_bf);
    // 2) qv = xn @ Wqkv + bqkv     (16384x512)
    gemm_bf16_kernel<false><<<1024, 256, 0, stream>>>(xn_bf, 256, wqkv_t,
                                                      bqkv, nullptr, qv, 512,
                                                      ROWS, 512, 256);
    // 3) split + hardswish(v) image
    post_qkv_kernel<<<ROWS, 256, 0, stream>>>(qv, q_bf, v_bf, hsv);
    // 4) 3x3 depthwise on hardswish(v)
    dwconv_v_kernel<<<ROWS, 256, 0, stream>>>(hsv, convpw, convpb, vconv);
    // 5) cross-shaped attention (horizontal then vertical halves)
    attn_kernel<<<dim3(32, 128), 64, 0, stream>>>(q_bf, v_bf, attn, 1);
    attn_kernel<<<dim3(32, 128), 64, 0, stream>>>(q_bf, v_bf, attn, 0);
    // 6) a = hardswish(attn + vconv)
    combine_kernel<<<ROWS, 256, 0, stream>>>(attn, vconv, a_bf);
    // 7) aout = a @ Wout + bout
    gemm_bf16_kernel<false><<<512, 256, 0, stream>>>(a_bf, 256, wout_t,
                                                     bout, nullptr, aout, 256,
                                                     ROWS, 256, 256);
    // 8) BatchNorm stats (training mode)
    bnstats_kernel<<<256, 256, 0, stream>>>(aout, mu, rvar);
    // 9) BN-apply + DES shortcut + residual -> x1
    bn_des_res_kernel<<<ROWS, 256, 0, stream>>>(x, aout, mu, rvar, bn_g, bn_b,
                                                des_wr, des_br, des_wl, des_bl, x1);
    // 10) LN2
    layernorm_bf16_kernel<<<ROWS, 256, 0, stream>>>(x1, ln2_w, ln2_b, xn2_bf);
    // 11) h1 = xn2 @ fc1_w + fc1_b   (16384x1024)
    gemm_bf16_kernel<false><<<2048, 256, 0, stream>>>(xn2_bf, 256, fc1_t,
                                                      fc1_b, nullptr, h1, 1024,
                                                      ROWS, 1024, 256);
    // 12) mix dwconvs + GELU -> bf16
    mixconv_kernel<<<65536, 256, 0, stream>>>(h1, mix3_w, mix3_b, mix5_w, mix5_b, hc_bf);
    // 13) out = hc @ fc2_w + fc2_b + x1
    gemm_bf16_kernel<true><<<512, 256, 0, stream>>>(hc_bf, 1024, fc2_t,
                                                    fc2_b, x1, out, 256,
                                                    ROWS, 256, 1024);

    (void)in_sizes; (void)n_in; (void)out_size; (void)ws_size;
}
